// SimplifiedLinearAttention_65180423685302
// MI455X (gfx1250) — matrix operand order repaired and checked by reading, not compiled, not run on hardware
//
#include <hip/hip_runtime.h>
#include <hip/hip_bf16.h>

// ---------------------------------------------------------------------------
// SimplifiedLinearAttention for MI455X (gfx1250), wave32 + WMMA f16->f32.
//
// Shapes: B=2048, N=256 (16x16 window), C=128, H=4, d=32, BH=8192.
// FLOPs: qkv 51.5G + proj 17.2G + linattn 8.6G + conv 3.4G ~= 81 GFLOP.
// HBM floor (x in + out): 512 MB -> ~22us @ 23.3 TB/s; GEMMs must use
// v_wmma_f32_16x16x32_f16 to keep compute under the memory roof.
// Intermediates kept in f16 in workspace (4 x 128 MB = 512 MB of d_ws).
//
// CDNA5 WMMA fragment layouts used (ISA 7.12.2, wave32):
//   A (16x32 f16):  lane l holds row m=l&15; elements 0-7 = K in
//                   [8*(l>>4), 8*(l>>4)+8), elements 8-15 = K in
//                   [16+8*(l>>4), ...+8)  -> two contiguous 16B chunks.
//   B (32x16 f16):  lane l holds col n=l&15; elements 0-7 = K in
//                   [8*(l>>4), ...+8), elements 8-15 = K in [16+8*(l>>4), ...+8) -> two 16B chunks.
//   D (16x16 f32):  vgpr r, lane l -> m = r + 8*(l>>4), n = l&15.
// ---------------------------------------------------------------------------

typedef __attribute__((ext_vector_type(16))) _Float16 v16h;
typedef __attribute__((ext_vector_type(8)))  _Float16 v8h;
typedef __attribute__((ext_vector_type(4)))  _Float16 v4h;
typedef __attribute__((ext_vector_type(8)))  float    v8f;

#define BQ    2048
#define NSEQ  256
#define CDIM  128
#define NHEAD 4
#define HD    32
#define BH    (BQ * NHEAD)      // 8192
#define HSZ   (NSEQ * HD)       // 8192 elems per (bh) head-plane

__device__ __forceinline__ v8f wmma_f16(v16h a, v16h b, v8f c) {
  // v_wmma_f32_16x16x32_f16, args: (neg_a, A, neg_b, B, c_mod, C, reuse_a, reuse_b)
  return __builtin_amdgcn_wmma_f32_16x16x32_f16(false, a, false, b, (short)0, c,
                                                false, false);
}

// A-fragment: row = caller-supplied row pointer (row-major, any stride).
__device__ __forceinline__ v16h frag_a(const _Float16* row, int kbase, int half) {
  union { v16h v; v8h h[2]; } u;
  u.h[0] = *(const v8h*)(row + kbase + 8 * half);
  u.h[1] = *(const v8h*)(row + kbase + 16 + 8 * half);
  return u.v;
}

// B-fragment: colrow = pointer to [n][k] row for this lane's column n.
__device__ __forceinline__ v16h frag_b(const _Float16* colrow, int kbase, int half) {
  union { v16h v; v8h h[2]; } u;
  const _Float16* p = colrow + kbase + 8 * half;
  u.h[0] = *(const v8h*)(p);
  u.h[1] = *(const v8h*)(p + 16);
  return u.v;
}

// ---------------------------------------------------------------------------
// Kernel 1: qkv = x @ qkv_w.T + b, fused epilogue:
//   q_p = relu(q)+1            -> qp  [bh][n][d]   (f16)
//   k_p = relu(k+pos_enc)+1    -> kpT [bh][d][n]   (f16, transposed)
//   v                          -> vT  [bh][d][n]   (f16, transposed = NCHW)
// Grid: (M/128, 3). Block 256 threads = 8 waves, tile 128x128, K=128.
// ---------------------------------------------------------------------------
__global__ __launch_bounds__(256) void qkv_kernel(
    const float* __restrict__ x, const float* __restrict__ qkv_w,
    const float* __restrict__ qkv_b, const float* __restrict__ pos_enc,
    _Float16* __restrict__ qp, _Float16* __restrict__ kpT,
    _Float16* __restrict__ vT) {
  __shared__ __align__(16) _Float16 xs[128 * 128];   // 32 KB
  __shared__ __align__(16) _Float16 wsh[128 * 128];  // 32 KB
  const int mt = blockIdx.x;   // 0..4095 row tiles
  const int nt = blockIdx.y;   // 0:q 1:k 2:v column section
  const int t  = threadIdx.x;
  const long rowbase = (long)mt * 128;

  // Stage x tile (fp32 -> f16) and this section's weight rows into LDS.
  const float* xsrc = x + rowbase * CDIM;
  const float* wsrc = qkv_w + (long)nt * 128 * CDIM;
#pragma unroll
  for (int it = 0; it < 16; ++it) {
    int f = it * 256 + t;  // float4 index, 4096 covers 128x128
    float4 dx = ((const float4*)xsrc)[f];
    float4 dw = ((const float4*)wsrc)[f];
    v4h hx = {(_Float16)dx.x, (_Float16)dx.y, (_Float16)dx.z, (_Float16)dx.w};
    v4h hw = {(_Float16)dw.x, (_Float16)dw.y, (_Float16)dw.z, (_Float16)dw.w};
    *(v4h*)(xs + f * 4)  = hx;
    *(v4h*)(wsh + f * 4) = hw;
  }
  __syncthreads();

  const int wid = t >> 5, lane = t & 31;
  const int wr = wid >> 1;       // 4 row groups of 32
  const int wc = wid & 1;        // 2 col groups of 64
  const int half = lane >> 4;

  v8f acc[2][4] = {};
#pragma unroll
  for (int ks = 0; ks < 4; ++ks) {
    const int kb = ks * 32;
    v16h a0 = frag_a(xs + (wr * 32 + (lane & 15)) * 128, kb, half);
    v16h a1 = frag_a(xs + (wr * 32 + 16 + (lane & 15)) * 128, kb, half);
#pragma unroll
    for (int j = 0; j < 4; ++j) {
      v16h b = frag_b(wsh + (wc * 64 + j * 16 + (lane & 15)) * 128, kb, half);
      acc[0][j] = wmma_f16(a0, b, acc[0][j]);
      acc[1][j] = wmma_f16(a1, b, acc[1][j]);
    }
  }

  // Epilogue: per D-layout, lane writes 8 rows of one column per tile.
#pragma unroll
  for (int i = 0; i < 2; ++i) {
#pragma unroll
    for (int j = 0; j < 4; ++j) {
      const int cl = wc * 64 + j * 16 + (lane & 15);  // 0..127 in section
      const int c  = nt * 128 + cl;
      const float bias = qkv_b[c];
      const int h = cl >> 5, d = cl & 31;
#pragma unroll
      for (int r = 0; r < 8; ++r) {
        const int mloc = wr * 32 + i * 16 + r + 8 * half;
        const long R = rowbase + mloc;
        const int b = (int)(R >> 8), n = (int)(R & 255);
        const long bh = (long)b * NHEAD + h;
        float val = acc[i][j][r] + bias;
        if (nt == 0) {                       // q -> q_p = relu(q)+1
          val = fmaxf(val, 0.f) + 1.f;
          qp[bh * HSZ + n * HD + d] = (_Float16)val;
        } else if (nt == 1) {                // k -> k_p = relu(k+pos)+1
          val += pos_enc[n * CDIM + cl];
          val = fmaxf(val, 0.f) + 1.f;
          kpT[bh * HSZ + d * NSEQ + n] = (_Float16)val;
        } else {                             // v (transposed = NCHW plane)
          vT[bh * HSZ + d * NSEQ + n] = (_Float16)val;
        }
      }
    }
  }
}

// ---------------------------------------------------------------------------
// Kernel 2: one workgroup per bh (8192 blocks, 256 threads = 8 waves).
//   kv = k_p^T v (32x32, K=256)   : split-K WMMA across 8 waves + LDS reduce
//   k_sum, normalizer             : VALU
//   x_attn = q_p kv (256x32,K=32) : WMMA, q_p A-frags straight from global
//   fv = depthwise 5x5 on vT      : VALU from LDS
//   att[b][n][h*32+d] = x_attn/(norm+1e-6) + fv + dwc_b   (f16)
// LDS hand-aliased: 56.7 KB total.
// ---------------------------------------------------------------------------
__global__ __launch_bounds__(256) void attn_kernel(
    const _Float16* __restrict__ qp, const _Float16* __restrict__ kpT,
    const _Float16* __restrict__ vT, const float* __restrict__ dwc_w,
    const float* __restrict__ dwc_b, _Float16* __restrict__ att) {
  __shared__ __align__(16) unsigned char smem[56704];
  _Float16* vTs   = (_Float16*)(smem);              // 16 KB, persists
  _Float16* kpTs  = (_Float16*)(smem + 16384);      // 16 KB, phase 1
  float*    kvprt = (float*)(smem + 32768);         //  8 KB, phase 1
  float*    atts  = (float*)(smem + 16384);         // 32 KB, phase 2 (aliases)
  _Float16* kvT16 = (_Float16*)(smem + 49152);      //  2 KB
  float*    normv = (float*)(smem + 51200);         //  1 KB
  float*    ksum  = (float*)(smem + 52224);         // 128 B
  float*    kspart= (float*)(smem + 52352);         //  1 KB
  float*    dwcw  = (float*)(smem + 53376);         // 3.2 KB
  float*    dwcb  = (float*)(smem + 56576);         // 128 B

  const int t = threadIdx.x;
  const long bh = blockIdx.x;
  const _Float16* kpg = kpT + bh * HSZ;
  const _Float16* vg  = vT + bh * HSZ;
  const _Float16* qbase = qp + bh * HSZ;

  // Phase 0: stage kpT / vT head-planes (16 KB each) + conv weights.
#pragma unroll
  for (int i = 0; i < 4; ++i) {
    ((uint4*)kpTs)[t + i * 256] = ((const uint4*)kpg)[t + i * 256];
    ((uint4*)vTs)[t + i * 256]  = ((const uint4*)vg)[t + i * 256];
  }
  for (int i = t; i < HD * 25; i += 256) dwcw[i] = dwc_w[i];
  if (t < HD) dwcb[t] = dwc_b[t];
  __syncthreads();

  // Phase 1a: k_sum partials (each thread sums 32 n for one d).
  {
    const int d = t >> 3, seg = t & 7;
    const _Float16* row = kpTs + d * NSEQ + seg * 32;
    float s = 0.f;
#pragma unroll
    for (int i = 0; i < 32; ++i) s += (float)row[i];
    kspart[t] = s;
  }
  // Phase 1b: kv WMMA. wave = (combo in 2x2 tiles) x (K half).
  {
    const int wid = t >> 5, lane = t & 31, half = lane >> 4;
    const int combo = wid & 3, khalf = wid >> 2;
    const int mt2 = combo >> 1, nt2 = combo & 1;
    v8f acc = {};
#pragma unroll
    for (int ks = 0; ks < 4; ++ks) {
      const int kb = khalf * 128 + ks * 32;
      v16h a = frag_a(kpTs + (mt2 * 16 + (lane & 15)) * NSEQ, kb, half);  // A=k_p^T
      v16h b = frag_b(vTs + (nt2 * 16 + (lane & 15)) * NSEQ, kb, half);   // B=v
      acc = wmma_f16(a, b, acc);
    }
#pragma unroll
    for (int r = 0; r < 8; ++r)
      kvprt[wid * 256 + (r + 8 * half) * 16 + (lane & 15)] = acc[r];
  }
  __syncthreads();

  // Phase 2: finalize ksum; reduce split-K kv and store transposed f16
  // (kvT16[d_out][d_in], i.e. B-operand layout for x_attn).
  if (t < HD) {
    float s = 0.f;
#pragma unroll
    for (int i = 0; i < 8; ++i) s += kspart[t * 8 + i];
    ksum[t] = s;
  }
#pragma unroll
  for (int c = 0; c < 4; ++c) {
    const float s = kvprt[c * 256 + t] + kvprt[(c + 4) * 256 + t];
    const int din = (c >> 1) * 16 + (t >> 4), dout = (c & 1) * 16 + (t & 15);
    kvT16[dout * HD + din] = (_Float16)s;
  }
  __syncthreads();

  // Phase 3: normalizer (thread t = position n), q_p row from global (64 B).
  {
    const _Float16* qrow = qbase + t * HD;
    float s = 0.f;
#pragma unroll
    for (int d = 0; d < HD; ++d) s += (float)qrow[d] * ksum[d];
    normv[t] = s + 1e-6f;
  }
  __syncthreads();  // kpTs/kvprt dead -> atts region live

  // Phase 4: x_attn = q_p @ kv. wave w owns rows [32w, 32w+32), K=32 = 1 WMMA.
  {
    const int wid = t >> 5, lane = t & 31, half = lane >> 4;
#pragma unroll
    for (int i = 0; i < 2; ++i) {
      const int rowb = wid * 32 + i * 16;
      v16h a = frag_a(qbase + (rowb + (lane & 15)) * HD, 0, half);
#pragma unroll
      for (int j = 0; j < 2; ++j) {
        v16h b = frag_b(kvT16 + (j * 16 + (lane & 15)) * HD, 0, half);
        v8f acc = {};
        acc = wmma_f16(a, b, acc);
#pragma unroll
        for (int r = 0; r < 8; ++r)
          atts[(rowb + r + 8 * half) * HD + j * 16 + (lane & 15)] = acc[r];
      }
    }
  }
  __syncthreads();

  // Phase 5: depthwise 5x5 conv on vTs (= NCHW plane) + combine + writeout.
  {
    const int d = t >> 3, p8 = t & 7;      // 8 threads per channel, 2 rows each
    const long b = bh >> 2;
    const int h = (int)(bh & 3);
    const _Float16* vrow = vTs + d * NSEQ;
    float wv[25];
#pragma unroll
    for (int i = 0; i < 25; ++i) wv[i] = dwcw[d * 25 + i];
    const float cb = dwcb[d];
#pragma unroll
    for (int py = 0; py < 2; ++py) {
      const int y = p8 * 2 + py;
      for (int xx0 = 0; xx0 < 16; ++xx0) {
        float a = cb;
#pragma unroll
        for (int ky = 0; ky < 5; ++ky) {
          const int yy = y + ky - 2;
          if (yy < 0 || yy >= 16) continue;
#pragma unroll
          for (int kx = 0; kx < 5; ++kx) {
            const int xx = xx0 + kx - 2;
            if (xx < 0 || xx >= 16) continue;
            a += (float)vrow[yy * 16 + xx] * wv[ky * 5 + kx];
          }
        }
        const int n = y * 16 + xx0;
        const float o = atts[n * HD + d] / normv[n] + a;
        att[((b * NSEQ + n) * CDIM) + h * HD + d] = (_Float16)o;
      }
    }
  }
}

// ---------------------------------------------------------------------------
// Kernel 3: out = att @ proj_w.T + proj_b  (f16 x f16 -> fp32 out).
// Grid: M/128 blocks; block 256 threads, tile 128x128, K=128.
// ---------------------------------------------------------------------------
__global__ __launch_bounds__(256) void proj_kernel(
    const _Float16* __restrict__ att, const float* __restrict__ proj_w,
    const float* __restrict__ proj_b, float* __restrict__ out) {
  __shared__ __align__(16) _Float16 as_[128 * 128];  // 32 KB
  __shared__ __align__(16) _Float16 wsh[128 * 128];  // 32 KB
  const long mt = blockIdx.x;
  const int t = threadIdx.x;
  const _Float16* asrc = att + mt * 128 * CDIM;
#pragma unroll
  for (int i = 0; i < 8; ++i)
    ((uint4*)as_)[t + i * 256] = ((const uint4*)asrc)[t + i * 256];
#pragma unroll
  for (int it = 0; it < 16; ++it) {
    int f = it * 256 + t;
    float4 dw = ((const float4*)proj_w)[f];
    v4h hw = {(_Float16)dw.x, (_Float16)dw.y, (_Float16)dw.z, (_Float16)dw.w};
    *(v4h*)(wsh + f * 4) = hw;
  }
  __syncthreads();

  const int wid = t >> 5, lane = t & 31, half = lane >> 4;
  const int wr = wid >> 1, wc = wid & 1;
  v8f acc[2][4] = {};
#pragma unroll
  for (int ks = 0; ks < 4; ++ks) {
    const int kb = ks * 32;
    v16h a0 = frag_a(as_ + (wr * 32 + (lane & 15)) * 128, kb, half);
    v16h a1 = frag_a(as_ + (wr * 32 + 16 + (lane & 15)) * 128, kb, half);
#pragma unroll
    for (int j = 0; j < 4; ++j) {
      v16h b = frag_b(wsh + (wc * 64 + j * 16 + (lane & 15)) * 128, kb, half);
      acc[0][j] = wmma_f16(a0, b, acc[0][j]);
      acc[1][j] = wmma_f16(a1, b, acc[1][j]);
    }
  }
#pragma unroll
  for (int i = 0; i < 2; ++i) {
#pragma unroll
    for (int j = 0; j < 4; ++j) {
      const int c = wc * 64 + j * 16 + (lane & 15);
      const float bias = proj_b[c];
#pragma unroll
      for (int r = 0; r < 8; ++r) {
        const long R = mt * 128 + wr * 32 + i * 16 + r + 8 * half;
        out[R * CDIM + c] = acc[i][j][r] + bias;
      }
    }
  }
}

// ---------------------------------------------------------------------------
extern "C" void kernel_launch(void* const* d_in, const int* in_sizes, int n_in,
                              void* d_out, int out_size, void* d_ws,
                              size_t ws_size, hipStream_t stream) {
  (void)in_sizes; (void)n_in; (void)out_size; (void)ws_size;
  const float* x      = (const float*)d_in[0];
  const float* qkv_w  = (const float*)d_in[1];
  const float* qkv_b  = (const float*)d_in[2];
  const float* pos    = (const float*)d_in[3];
  const float* dwc_w  = (const float*)d_in[4];
  const float* dwc_b  = (const float*)d_in[5];
  const float* proj_w = (const float*)d_in[6];
  const float* proj_b = (const float*)d_in[7];
  float* out = (float*)d_out;

  // Workspace: 4 f16 tensors of BH*N*D = 128 MB each (512 MB total).
  const size_t PLANE = (size_t)BH * HSZ * sizeof(_Float16);  // 128 MB
  char* ws = (char*)d_ws;
  _Float16* qp  = (_Float16*)(ws);
  _Float16* kpT = (_Float16*)(ws + PLANE);
  _Float16* vT  = (_Float16*)(ws + 2 * PLANE);
  _Float16* att = (_Float16*)(ws + 3 * PLANE);

  dim3 gA(4096, 3);  // (M/128 row tiles) x (q,k,v sections)
  qkv_kernel<<<gA, 256, 0, stream>>>(x, qkv_w, qkv_b, pos, qp, kpT, vT);
  attn_kernel<<<BH, 256, 0, stream>>>(qp, kpT, vT, dwc_w, dwc_b, att);
  proj_kernel<<<4096, 256, 0, stream>>>(att, proj_w, proj_b, out);
}
